// HalfKPNNUE_18287970746445
// MI455X (gfx1250) — matrix operand order repaired and checked by reading, not compiled, not run on hardware
//
#include <hip/hip_runtime.h>
#include <hip/hip_bf16.h>
#include <math.h>

#define INPUT_SIZE 40960
#define HIDDEN 256
#define MAX_FEATS 32
#define BATCH 8192

typedef _Float16 v16h __attribute__((ext_vector_type(16)));
typedef float    v8f  __attribute__((ext_vector_type(8)));

union Frag { v16h v; uint4 q[2]; };
union H8   { uint4 q; _Float16 x[8]; };

// ---------------------------------------------------------------------------
// f32 -> f16 conversion (for w1, w2)
// ---------------------------------------------------------------------------
__global__ void k_cvt_f16(const float* __restrict__ in, _Float16* __restrict__ out, int n) {
  int i = blockIdx.x * blockDim.x + threadIdx.x;
  if (i < n) out[i] = (_Float16)in[i];
}

// ---------------------------------------------------------------------------
// Transpose + convert ft_W f32 [256][40960] -> ftWt f16 [40960][256].
// Halves the dominant L2 gather traffic of the accumulate phase.
// grid (1280, 8), block (32, 8)
// ---------------------------------------------------------------------------
__global__ void k_transpose_ftW(const float* __restrict__ W, _Float16* __restrict__ Wt) {
  __shared__ float tile[32][33];
  const int tx = threadIdx.x, ty = threadIdx.y;
  const int x  = blockIdx.x * 32 + tx;   // input-feature dim (40960)
  const int y0 = blockIdx.y * 32;        // hidden dim (256)
#pragma unroll
  for (int i = ty; i < 32; i += 8)
    tile[i][tx] = W[(size_t)(y0 + i) * INPUT_SIZE + x];
  __syncthreads();
  const int xo  = y0 + tx;               // hidden now contiguous
  const int yo0 = blockIdx.x * 32;
#pragma unroll
  for (int i = ty; i < 32; i += 8)
    Wt[(size_t)(yo0 + i) * HIDDEN + xo] = (_Float16)tile[tx][i];
}

// ---------------------------------------------------------------------------
// Sparse accumulate + clip + stm select -> xh f16 [BATCH][512].
// One wave per batch item; lane owns hidden elems 8*lane..8*lane+7, so every
// feature gather is a single b128 load (512B contiguous per wave). Feature
// index is wave-uniform -> uniform branch. Accumulation in f32 registers.
// grid BATCH/8, block 256 (8 waves = 8 batch items).
// ---------------------------------------------------------------------------
__global__ void k_accum(const int* __restrict__ wf, const int* __restrict__ bf,
                        const float* __restrict__ stm,
                        const _Float16* __restrict__ ftWt, const float* __restrict__ ft_b,
                        _Float16* __restrict__ xh) {
  const int lane = threadIdx.x & 31;
  const int b    = blockIdx.x * 8 + (threadIdx.x >> 5);
  const int e0   = lane * 8;

  float wacc[8], kacc[8];
#pragma unroll
  for (int j = 0; j < 8; ++j) { wacc[j] = ft_b[e0 + j]; kacc[j] = wacc[j]; }

  const int* __restrict__ wrow = wf + (size_t)b * MAX_FEATS;
  const int* __restrict__ brow = bf + (size_t)b * MAX_FEATS;
#pragma unroll 4
  for (int i = 0; i < MAX_FEATS; ++i) {
    int iw = wrow[i];
    if (iw >= 0) {
      H8 g; g.q = *(const uint4*)(ftWt + (size_t)iw * HIDDEN + e0);
#pragma unroll
      for (int j = 0; j < 8; ++j) wacc[j] += (float)g.x[j];
    }
    int ib = brow[i];
    if (ib >= 0) {
      H8 g; g.q = *(const uint4*)(ftWt + (size_t)ib * HIDDEN + e0);
#pragma unroll
      for (int j = 0; j < 8; ++j) kacc[j] += (float)g.x[j];
    }
  }

  const float s = stm[b], r = 1.0f - s;
  H8 us, th;
#pragma unroll
  for (int j = 0; j < 8; ++j) {
    float w = fminf(fmaxf(wacc[j], 0.f), 127.f);
    float k = fminf(fmaxf(kacc[j], 0.f), 127.f);
    us.x[j] = (_Float16)(s * w + r * k);
    th.x[j] = (_Float16)(s * k + r * w);
  }
  *(uint4*)(xh + (size_t)b * 512 + e0)       = us.q;
  *(uint4*)(xh + (size_t)b * 512 + 256 + e0) = th.q;
}

// ---------------------------------------------------------------------------
// Layer 1 via WMMA: x[8192,512] @ w1^T[512,32] + b1, ReLU -> x1h f16 [8192][32]
// w1 (32KB f16) is staged into LDS once per block with the gfx1250 async
// global->LDS copy; all 8 waves read B fragments from LDS (ds_load_b128).
// A/B fragments use k = 16h..16h+15 per lane (two contiguous 16B pieces) -> b128 loads.
// grid 64, block 256 (8 waves, one 16-row batch tile each).
// ---------------------------------------------------------------------------
__global__ void k_layer1_wmma(const _Float16* __restrict__ xh,
                              const _Float16* __restrict__ w1h,
                              const float* __restrict__ b1,
                              _Float16* __restrict__ x1h) {
  __shared__ _Float16 w1s[32 * 512];   // 32 KB

  // --- async stage w1h -> LDS (2048 x b128, 8 per thread) ------------------
  {
    unsigned lbase = (unsigned)(unsigned long long)
        (__attribute__((address_space(3))) _Float16*)w1s;
    unsigned long long gbase = (unsigned long long)w1h;
    for (int i = threadIdx.x; i < 2048; i += 256) {
      unsigned           laddr = lbase + (unsigned)i * 16u;
      unsigned long long gaddr = gbase + (unsigned long long)i * 16ull;
      asm volatile("global_load_async_to_lds_b128 %0, %1, off"
                   :: "v"(laddr), "v"(gaddr) : "memory");
    }
    asm volatile("s_wait_asynccnt 0x0" ::: "memory");
  }
  __syncthreads();

  const int lane = threadIdx.x & 31;
  const int wave = blockIdx.x * (blockDim.x >> 5) + (threadIdx.x >> 5);
  const int batchBase = wave * 16;
  const int m = lane & 15;   // A row / D column index
  const int h = lane >> 4;   // K-half select

  const _Float16* __restrict__ arow = xh + (size_t)(batchBase + m) * 512;

  v8f c0 = {}, c1 = {};
  for (int c = 0; c < 16; ++c) {
    const int kbase = 32 * c;
    Frag A, B0, B1;
    // A (16x32 f16): VGPRs 0-7 = K 16h..16h+15
    A.q[0] = *(const uint4*)(arow + kbase + 16 * h);
    A.q[1] = *(const uint4*)(arow + kbase + 16 * h + 8);
    // B (32x16 f16): column n = lane&15 ; VGPRs 0-7 = K 16h..16h+15
    const uint4* b0p = (const uint4*)(w1s + (size_t)m * 512 + kbase + 16 * h);
    const uint4* b1p = (const uint4*)(w1s + (size_t)(16 + m) * 512 + kbase + 16 * h);
    B0.q[0] = b0p[0]; B0.q[1] = b0p[1];
    B1.q[0] = b1p[0]; B1.q[1] = b1p[1];
    c0 = __builtin_amdgcn_wmma_f32_16x16x32_f16(false, A.v, false, B0.v, (short)0, c0, false, false);
    c1 = __builtin_amdgcn_wmma_f32_16x16x32_f16(false, A.v, false, B1.v, (short)0, c1, false, false);
  }

  const float bias0 = b1[m];
  const float bias1 = b1[16 + m];
#pragma unroll
  for (int r = 0; r < 8; ++r) {
    const int row = batchBase + r + 8 * h;   // D: M = r + 8*(lane>>4), N = lane&15
    float v0 = c0[r] + bias0; v0 = v0 > 0.f ? v0 : 0.f;
    float v1 = c1[r] + bias1; v1 = v1 > 0.f ? v1 : 0.f;
    x1h[(size_t)row * 32 + m]      = (_Float16)v0;
    x1h[(size_t)row * 32 + 16 + m] = (_Float16)v1;
  }
}

// ---------------------------------------------------------------------------
// Layer 2 via WMMA (K=32, one WMMA per N-tile) + ReLU, fused with the 32->1
// output dot + sigmoid. Half-wave shfl_xor reduction matches the D layout
// (lanes 0-15 hold rows M=r, lanes 16-31 hold M=r+8).
// grid 64, block 256.
// ---------------------------------------------------------------------------
__global__ void k_layer23_wmma(const _Float16* __restrict__ x1h,
                               const _Float16* __restrict__ w2h,
                               const float* __restrict__ b2,
                               const float* __restrict__ wo,
                               const float* __restrict__ bo,
                               float* __restrict__ out) {
  const int lane = threadIdx.x & 31;
  const int wave = blockIdx.x * (blockDim.x >> 5) + (threadIdx.x >> 5);
  const int batchBase = wave * 16;
  const int m = lane & 15;
  const int h = lane >> 4;

  Frag A, B0, B1;
  const _Float16* __restrict__ arow = x1h + (size_t)(batchBase + m) * 32;
  A.q[0] = *(const uint4*)(arow + 16 * h);
  A.q[1] = *(const uint4*)(arow + 16 * h + 8);
  const uint4* b0p = (const uint4*)(w2h + (size_t)m * 32 + 16 * h);
  const uint4* b1p = (const uint4*)(w2h + (size_t)(16 + m) * 32 + 16 * h);
  B0.q[0] = b0p[0]; B0.q[1] = b0p[1];
  B1.q[0] = b1p[0]; B1.q[1] = b1p[1];

  v8f c0 = {}, c1 = {};
  c0 = __builtin_amdgcn_wmma_f32_16x16x32_f16(false, A.v, false, B0.v, (short)0, c0, false, false);
  c1 = __builtin_amdgcn_wmma_f32_16x16x32_f16(false, A.v, false, B1.v, (short)0, c1, false, false);

  const float bias0 = b2[m], bias1 = b2[16 + m];
  const float wo0 = wo[m], wo1 = wo[16 + m];
  const float bias_o = bo[0];
#pragma unroll
  for (int r = 0; r < 8; ++r) {
    float v0 = c0[r] + bias0; v0 = v0 > 0.f ? v0 : 0.f;
    float v1 = c1[r] + bias1; v1 = v1 > 0.f ? v1 : 0.f;
    float v = v0 * wo0 + v1 * wo1;
    // reduce across each 16-lane half (masks < 16 never cross halves)
    v += __shfl_xor(v, 1);
    v += __shfl_xor(v, 2);
    v += __shfl_xor(v, 4);
    v += __shfl_xor(v, 8);
    float z = v + bias_o;
    float sig = 1.0f / (1.0f + expf(-z));
    if (m == 0)                      // lanes 0 and 16: rows r and r+8
      out[batchBase + r + 8 * h] = sig;
  }
}

// ---------------------------------------------------------------------------
// Launch
// ---------------------------------------------------------------------------
extern "C" void kernel_launch(void* const* d_in, const int* in_sizes, int n_in,
                              void* d_out, int out_size, void* d_ws, size_t ws_size,
                              hipStream_t stream) {
  const int*   wf   = (const int*)  d_in[0];
  const int*   bf   = (const int*)  d_in[1];
  const float* stm  = (const float*)d_in[2];
  const float* ft_W = (const float*)d_in[3];
  const float* ft_b = (const float*)d_in[4];
  const float* w1   = (const float*)d_in[5];
  const float* b1   = (const float*)d_in[6];
  const float* w2   = (const float*)d_in[7];
  const float* b2   = (const float*)d_in[8];
  const float* wo   = (const float*)d_in[9];
  const float* bo   = (const float*)d_in[10];
  float* out = (float*)d_out;

  // workspace carve-up (all offsets 256B-aligned)
  char* base = (char*)d_ws;
  _Float16* ftWt = (_Float16*)(base);                        // 40960*256*2 = 20,971,520
  _Float16* xh   = (_Float16*)(base + 20971520);             // 8192*512*2  =  8,388,608
  _Float16* x1h  = (_Float16*)(base + 20971520 + 8388608);   // 8192*32*2   =    524,288
  _Float16* w1h  = (_Float16*)(base + 20971520 + 8388608 + 524288);          // 32,768
  _Float16* w2h  = (_Float16*)(base + 20971520 + 8388608 + 524288 + 32768);  //  2,048

  // 1) weight conversions (independent)
  k_cvt_f16<<<(32 * 512 + 255) / 256, 256, 0, stream>>>(w1, w1h, 32 * 512);
  k_cvt_f16<<<(32 * 32 + 255) / 256, 256, 0, stream>>>(w2, w2h, 32 * 32);

  // 2) transpose + f16-convert feature table for coalesced half-size gathers
  k_transpose_ftW<<<dim3(INPUT_SIZE / 32, HIDDEN / 32), dim3(32, 8), 0, stream>>>(ft_W, ftWt);

  // 3) sparse accumulate + clip + stm select
  k_accum<<<BATCH / 8, 256, 0, stream>>>(wf, bf, stm, ftWt, ft_b, xh);

  // 4) layer 1 (512->32) via WMMA with async-LDS-staged weights
  k_layer1_wmma<<<BATCH / (16 * 8), 256, 0, stream>>>(xh, w1h, b1, x1h);

  // 5) layer 2 (32->32) via WMMA + output layer + sigmoid
  k_layer23_wmma<<<BATCH / (16 * 8), 256, 0, stream>>>(x1h, w2h, b2, wo, bo, out);
}
